// PolicyNetwork_23785528886023
// MI455X (gfx1250) — matrix operand order repaired and checked by reading, not compiled, not run on hardware
//
#include <hip/hip_runtime.h>

typedef __attribute__((ext_vector_type(16))) __bf16 v16bf;
typedef __attribute__((ext_vector_type(8)))  __bf16 v8bf;
typedef __attribute__((ext_vector_type(8)))  float  v8f;

#define Bn 256   // batch
#define Dd 256   // input dim
#define Hh 512   // hidden
#define G3 1536  // 3*H
#define Mm 512   // mlp dim
#define Aa 64    // actions

// ---------------------------------------------------------------- helpers

__device__ __forceinline__ float fast_sigmoid(float x) {
    return 1.0f / (1.0f + __expf(-x));
}
__device__ __forceinline__ float fast_tanh(float x) {
    float ax = fabsf(x);
    float t  = __expf(-2.0f * ax);
    float r  = (1.0f - t) / (1.0f + t);
    return copysignf(r, x);
}

// B fragment (32x16 bf16, K x N): weights pre-packed as [kt][col][32 k-elems].
// Lane L: column = colbase + (L&15); lanes 0-15 hold K 0..15, lanes 16-31 hold
// K 16..31 of this k-tile -> one contiguous 32B read per lane.
__device__ __forceinline__ v16bf load_bfrag(const __bf16* __restrict__ Wp,
                                            int kt, int colbase, int lane, int Gtot) {
    int n = colbase + (lane & 15);
    int h = lane >> 4;
    const __bf16* p = Wp + ((size_t)(kt * Gtot + n) * 32) + 16 * h;
    return *(const v16bf*)p;
}

// A fragment (16x32 bf16, M x K) from row-major bf16 [rows, lda].
// Lane L: row = rowbase + (L&15); VGPR0-7 = K[16h..16h+15].
__device__ __forceinline__ v16bf load_afrag_bf16(const __bf16* __restrict__ A,
                                                 int rowbase, int kbase, int lane, int lda) {
    int m = lane & 15;
    int h = lane >> 4;
    const __bf16* p = A + (size_t)(rowbase + m) * lda + kbase + 16 * h;
    v8bf c0 = *(const v8bf*)p;
    v8bf c1 = *(const v8bf*)(p + 8);
    v16bf a;
#pragma unroll
    for (int i = 0; i < 8; ++i) { a[i] = c0[i]; a[8 + i] = c1[i]; }
    return a;
}

// A fragment from row-major f32 source, converted to bf16 in-register.
__device__ __forceinline__ v16bf load_afrag_f32(const float* __restrict__ A,
                                                int rowbase, int kbase, int lane, int lda) {
    int m = lane & 15;
    int h = lane >> 4;
    const float* p = A + (size_t)(rowbase + m) * lda + kbase + 16 * h;
    float4 f0 = *(const float4*)(p);
    float4 f1 = *(const float4*)(p + 4);
    float4 f2 = *(const float4*)(p + 8);
    float4 f3 = *(const float4*)(p + 12);
    v16bf a;
    a[0]  = (__bf16)f0.x; a[1]  = (__bf16)f0.y; a[2]  = (__bf16)f0.z; a[3]  = (__bf16)f0.w;
    a[4]  = (__bf16)f1.x; a[5]  = (__bf16)f1.y; a[6]  = (__bf16)f1.z; a[7]  = (__bf16)f1.w;
    a[8]  = (__bf16)f2.x; a[9]  = (__bf16)f2.y; a[10] = (__bf16)f2.z; a[11] = (__bf16)f2.w;
    a[12] = (__bf16)f3.x; a[13] = (__bf16)f3.y; a[14] = (__bf16)f3.z; a[15] = (__bf16)f3.w;
    return a;
}

#define WMMA_BF16(A_, B_, C_) \
    __builtin_amdgcn_wmma_f32_16x16x32_bf16(false, (A_), false, (B_), (short)0, (C_), false, false)

// ---------------------------------------------------------------- prep kernels

// Pack f32 weights [G rows, K cols] row-major -> bf16 [ktOff + K/32][Gdst][32]
__global__ void pack_w(const float* __restrict__ src, __bf16* __restrict__ dst,
                       int G, int K, int ktOff, int Gdst) {
    int idx = blockIdx.x * blockDim.x + threadIdx.x;
    if (idx >= G * K) return;
    int g = idx / K;
    int k = idx - g * K;
    dst[((size_t)(ktOff + (k >> 5)) * Gdst + g) * 32 + (k & 31)] = (__bf16)src[idx];
}

__global__ void zero_h(float* __restrict__ hf, __bf16* __restrict__ hb) {
    int idx = blockIdx.x * blockDim.x + threadIdx.x;
    if (idx >= Bn * Hh) return;
    hf[idx] = 0.0f;
    hb[idx] = (__bf16)0.0f;
}

// ---------------------------------------------------------------- GRU step
// One launch per timestep. Fused: [x_t | h] @ [W_ih | W_hh]^T (K=768, bf16 WMMA,
// f32 accum, n-gate x/h parts kept separate) + gates + h update, all in registers.
// Block = 256 thr = 8 waves (2 rows x 4 cols of 16x16 tiles). Grid = (H/64, B/32).
__global__ void __launch_bounds__(256)
gru_step(const float* __restrict__ xt,
         const __bf16* __restrict__ h_in_bf, const float* __restrict__ h_in_f,
         __bf16* __restrict__ h_out_bf, float* __restrict__ h_out_f,
         const __bf16* __restrict__ Wp,
         const float* __restrict__ b_ih, const float* __restrict__ b_hh) {
    const int wave  = threadIdx.x >> 5;
    const int lane  = threadIdx.x & 31;
    const int bbase = blockIdx.y * 32 + (wave >> 2) * 16;   // batch rows
    const int jbase = blockIdx.x * 64 + (wave & 3) * 16;    // hidden cols

    v8f acc_r = {}, acc_z = {}, acc_nx = {}, acc_nh = {};

    // x-part: K tiles 0..7 (f32 -> bf16 convert on the fly)
#pragma unroll
    for (int kt = 0; kt < 8; ++kt) {
        v16bf a  = load_afrag_f32(xt, bbase, kt * 32, lane, Dd);
        v16bf br = load_bfrag(Wp, kt, jbase,            lane, G3);
        v16bf bz = load_bfrag(Wp, kt, Hh + jbase,       lane, G3);
        v16bf bn = load_bfrag(Wp, kt, 2 * Hh + jbase,   lane, G3);
        acc_r  = WMMA_BF16(a, br, acc_r);
        acc_z  = WMMA_BF16(a, bz, acc_z);
        acc_nx = WMMA_BF16(a, bn, acc_nx);
    }
    // h-part: K tiles 8..23 (bf16 h)
#pragma unroll
    for (int kt = 8; kt < 24; ++kt) {
        v16bf a  = load_afrag_bf16(h_in_bf, bbase, (kt - 8) * 32, lane, Hh);
        v16bf br = load_bfrag(Wp, kt, jbase,            lane, G3);
        v16bf bz = load_bfrag(Wp, kt, Hh + jbase,       lane, G3);
        v16bf bn = load_bfrag(Wp, kt, 2 * Hh + jbase,   lane, G3);
        acc_r  = WMMA_BF16(a, br, acc_r);
        acc_z  = WMMA_BF16(a, bz, acc_z);
        acc_nh = WMMA_BF16(a, bn, acc_nh);
    }

    // gate math + hidden-state update (C layout: VGPR v, lane L ->
    // row = bbase + v + 8*(L>>4), col = jbase + (L&15); identical for all accs)
    const int jcol = jbase + (lane & 15);
    const float br_b = b_ih[jcol]          + b_hh[jcol];
    const float bz_b = b_ih[Hh + jcol]     + b_hh[Hh + jcol];
    const float bn_x = b_ih[2 * Hh + jcol];
    const float bn_h = b_hh[2 * Hh + jcol];
#pragma unroll
    for (int v = 0; v < 8; ++v) {
        int row   = bbase + v + 8 * (lane >> 4);
        float hv  = h_in_f[(size_t)row * Hh + jcol];
        float r   = fast_sigmoid(acc_r[v] + br_b);
        float z   = fast_sigmoid(acc_z[v] + bz_b);
        float n   = fast_tanh(acc_nx[v] + bn_x + r * (acc_nh[v] + bn_h));
        float hnw = (1.0f - z) * n + z * hv;
        h_out_f [(size_t)row * Hh + jcol] = hnw;
        h_out_bf[(size_t)row * Hh + jcol] = (__bf16)hnw;
    }
}

// ---------------------------------------------------------------- MLP GEMM+tanh
// out = tanh(A[256,K] @ Wp_packed[K,N] + bias). Block = 128 thr = 4 waves
// stacked in M; each wave does a 16x64 tile. Grid = (N/64, 256/64).
__global__ void __launch_bounds__(128)
wmma_gemm_tanh(const __bf16* __restrict__ A, const __bf16* __restrict__ Wp,
               const float* __restrict__ bias,
               __bf16* __restrict__ out_bf, float* __restrict__ out_f,
               int K, int N) {
    const int wave  = threadIdx.x >> 5;
    const int lane  = threadIdx.x & 31;
    const int rbase = blockIdx.y * 64 + wave * 16;
    const int cbase = blockIdx.x * 64;

    v8f acc[4] = {};
    for (int kt = 0; kt < (K >> 5); ++kt) {
        v16bf a = load_afrag_bf16(A, rbase, kt * 32, lane, K);
#pragma unroll
        for (int t = 0; t < 4; ++t) {
            v16bf b = load_bfrag(Wp, kt, cbase + 16 * t, lane, N);
            acc[t] = WMMA_BF16(a, b, acc[t]);
        }
    }
#pragma unroll
    for (int t = 0; t < 4; ++t) {
        int col  = cbase + 16 * t + (lane & 15);
        float bb = bias[col];
#pragma unroll
        for (int v = 0; v < 8; ++v) {
            int row   = rbase + v + 8 * (lane >> 4);
            float val = fast_tanh(acc[t][v] + bb);
            if (out_f)  out_f [(size_t)row * N + col] = val;
            if (out_bf) out_bf[(size_t)row * N + col] = (__bf16)val;
        }
    }
}

// ---------------------------------------------------------------- launcher

extern "C" void kernel_launch(void* const* d_in, const int* in_sizes, int n_in,
                              void* d_out, int out_size, void* d_ws, size_t ws_size,
                              hipStream_t stream) {
    (void)in_sizes; (void)n_in; (void)out_size; (void)ws_size;

    const float* x    = (const float*)d_in[0];  // [T,B,D]
    const float* W_ih = (const float*)d_in[1];  // [3H,D]
    const float* W_hh = (const float*)d_in[2];  // [3H,H]
    const float* b_ih = (const float*)d_in[3];  // [3H]
    const float* b_hh = (const float*)d_in[4];  // [3H]
    const float* W1   = (const float*)d_in[5];  // [M,H]
    const float* b1   = (const float*)d_in[6];  // [M]
    const float* W2   = (const float*)d_in[7];  // [A,M]
    const float* b2   = (const float*)d_in[8];  // [A]

    char*  ws  = (char*)d_ws;
    size_t off = 0;
    auto salloc = [&](size_t bytes) -> void* {
        void* p = ws + off;
        off = (off + bytes + 255) & ~(size_t)255;
        return p;
    };

    __bf16* Wp     = (__bf16*)salloc((size_t)24 * G3 * 32 * 2);  // fused [Wih|Whh] packed
    __bf16* W1p    = (__bf16*)salloc((size_t)16 * Mm * 32 * 2);
    __bf16* W2p    = (__bf16*)salloc((size_t)16 * Aa * 32 * 2);
    float*  hf0    = (float*) salloc((size_t)Bn * Hh * 4);
    float*  hf1    = (float*) salloc((size_t)Bn * Hh * 4);
    __bf16* hb0    = (__bf16*)salloc((size_t)Bn * Hh * 2);
    __bf16* hb1    = (__bf16*)salloc((size_t)Bn * Hh * 2);
    __bf16* hid_bf = (__bf16*)salloc((size_t)Bn * Mm * 2);

    // One-time weight packing (bf16, WMMA-B fragment layout)
    pack_w<<<(G3 * Dd + 255) / 256, 256, 0, stream>>>(W_ih, Wp, G3, Dd, 0, G3);
    pack_w<<<(G3 * Hh + 255) / 256, 256, 0, stream>>>(W_hh, Wp, G3, Hh, 8, G3);
    pack_w<<<(Mm * Hh + 255) / 256, 256, 0, stream>>>(W1, W1p, Mm, Hh, 0, Mm);
    pack_w<<<(Aa * Mm + 255) / 256, 256, 0, stream>>>(W2, W2p, Aa, Mm, 0, Aa);
    zero_h<<<(Bn * Hh + 255) / 256, 256, 0, stream>>>(hf0, hb0);

    // Sequential recurrence: one fused WMMA step kernel per timestep,
    // double-buffered h (even t reads buf0, writes buf1; final h lands in buf0).
    dim3 gGrid(Hh / 64, Bn / 32);
    for (int t = 0; t < 512; ++t) {
        const float* xt = x + (size_t)t * Bn * Dd;
        if ((t & 1) == 0)
            gru_step<<<gGrid, 256, 0, stream>>>(xt, hb0, hf0, hb1, hf1, Wp, b_ih, b_hh);
        else
            gru_step<<<gGrid, 256, 0, stream>>>(xt, hb1, hf1, hb0, hf0, Wp, b_ih, b_hh);
    }

    // MLP head: hidden = tanh(h@W1^T + b1); action = tanh(hidden@W2^T + b2)
    wmma_gemm_tanh<<<dim3(Mm / 64, Bn / 64), 128, 0, stream>>>(
        hb0, W1p, b1, hid_bf, (float*)nullptr, Hh, Mm);
    wmma_gemm_tanh<<<dim3(Aa / 64, Bn / 64), 128, 0, stream>>>(
        hid_bf, W2p, b2, (__bf16*)nullptr, (float*)d_out, Mm, Aa);
}
